// Block_16578573762685
// MI455X (gfx1250) — matrix operand order repaired and checked by reading, not compiled, not run on hardware
//
#include <hip/hip_runtime.h>
#include <hip/hip_bf16.h>

typedef __attribute__((ext_vector_type(16))) __bf16 v16bf;
typedef __attribute__((ext_vector_type(8)))  __bf16 v8bf;
typedef __attribute__((ext_vector_type(8)))  float  v8f;

#define DIMX  384
#define NHEAD 6
#define HDX   64
#define TSEQ  2048
#define BATCH 8
#define NTOK  (BATCH*TSEQ)
#define HDIMX 768

__device__ inline v8f wmma_bf16(v16bf a, v16bf b, v8f c) {
  // D = A(16x32 bf16) * B(32x16 bf16) + C(16x16 f32)
  return __builtin_amdgcn_wmma_f32_16x16x32_bf16(false, a, false, b, (short)0, c,
                                                 false, false);
}

__device__ inline float gelu_f(float v) {
  return 0.5f * v * (1.0f + erff(v * 0.70710678118654752f));
}

// ---------------------------------------------------------------------------
// Weight pre-pack: row-major f32 W[K,N] -> bf16 WMMA B-fragment layout.
// Fragment tile (kt,nt): 32 lanes x 16 elems; lane = {half=lane/16, n=lane%16},
// element e maps to W[kt*32 + 16*half + e][nt*16 + n].
// ---------------------------------------------------------------------------
__global__ __launch_bounds__(256) void pack_w_k(const float* __restrict__ W,
                                                __bf16* __restrict__ Wp,
                                                int K, int Nn) {
  int idx = blockIdx.x * 256 + threadIdx.x;
  if (idx >= K * Nn) return;
  int e    = idx & 15;
  int lane = (idx >> 4) & 31;
  int tile = idx >> 9;
  int nT   = Nn >> 4;
  int kt   = tile / nT;
  int nt   = tile - kt * nT;
  int k    = kt * 32 + (lane >> 4) * 16 + e;
  int n    = nt * 16 + (lane & 15);
  Wp[idx] = (__bf16)W[(size_t)k * Nn + n];
}

// ---------------------------------------------------------------------------
// RMSNorm: one wave per token (DIMX=384 = 12 floats/lane), bf16 output.
// ---------------------------------------------------------------------------
__global__ __launch_bounds__(256) void rmsnorm_k(const float* __restrict__ x,
                                                 const float* __restrict__ w,
                                                 __bf16* __restrict__ out) {
  const int tok  = blockIdx.x * 8 + (threadIdx.x >> 5);
  const int lane = threadIdx.x & 31;
  const float* xp = x + (size_t)tok * DIMX;
  float v[12];
  float ss = 0.f;
#pragma unroll
  for (int i = 0; i < 12; ++i) { v[i] = xp[lane + 32 * i]; ss += v[i] * v[i]; }
#pragma unroll
  for (int o = 16; o >= 1; o >>= 1) ss += __shfl_xor(ss, o, 32);
  const float inv = rsqrtf(ss * (1.0f / DIMX) + 1e-6f);
  __bf16* op = out + (size_t)tok * DIMX;
#pragma unroll
  for (int i = 0; i < 12; ++i)
    op[lane + 32 * i] = (__bf16)(v[i] * inv * w[lane + 32 * i]);
}

// ---------------------------------------------------------------------------
// WMMA GEMM: C[M,N] = A_bf16[M,K] @ Wpacked_bf16[K,N], epilogue variants.
// 8 waves/block; each wave computes a 32x64 strip: 2 A-fragments share the
// same 4 B-fragments -> 8 WMMAs per 32-wide K step.
// ---------------------------------------------------------------------------
enum { EPI_BF16 = 0, EPI_VT = 1, EPI_RESF32 = 2, EPI_GELU_BF16 = 3,
       EPI_BIAS_RES_F32 = 4 };

template <int EPI>
__global__ __launch_bounds__(256) void gemm_wmma(
    const __bf16* __restrict__ A, const __bf16* __restrict__ Wp,
    const float* __restrict__ bias, const float* __restrict__ res,
    __bf16* __restrict__ outB, float* __restrict__ outF,
    int M, int K, int N) {
  const int wave = threadIdx.x >> 5;
  const int lane = threadIdx.x & 31;
  const int half = lane >> 4;
  const int nl   = lane & 15;
  const int row0 = (blockIdx.x * 8 + wave) * 32;
  const int col0 = blockIdx.y * 64;
  const int nT16 = N >> 4;
  const v8f zero = {};
  v8f acc[2][4];
#pragma unroll
  for (int i = 0; i < 2; ++i)
#pragma unroll
    for (int j = 0; j < 4; ++j) acc[i][j] = zero;
  const int kSteps = K >> 5;
  // A layout (16-bit 16x32): lane m=nl; e0..15 -> K=16*half+e
  const __bf16* ap0 = A + (size_t)(row0 + nl) * K + half * 16;
  const size_t rstep = (size_t)16 * K;
  for (int kt = 0; kt < kSteps; ++kt) {
    union { v16bf v; v8bf p[2]; } a[2];
#pragma unroll
    for (int i = 0; i < 2; ++i) {
      const __bf16* ap = ap0 + i * rstep + kt * 32;
      a[i].p[0] = *(const v8bf*)(ap);
      a[i].p[1] = *(const v8bf*)(ap + 8);
    }
    const __bf16* wp = Wp + ((size_t)(kt * nT16 + (col0 >> 4)) << 9) + lane * 16;
#pragma unroll
    for (int j = 0; j < 4; ++j) {
      v16bf bf = *(const v16bf*)(wp + ((size_t)j << 9));
      acc[0][j] = wmma_bf16(a[0].v, bf, acc[0][j]);
      acc[1][j] = wmma_bf16(a[1].v, bf, acc[1][j]);
    }
  }
  // C layout: VGPR r -> row r + 8*half; col = nl.
#pragma unroll
  for (int i = 0; i < 2; ++i) {
    const int rowb = row0 + i * 16;
#pragma unroll
    for (int j = 0; j < 4; ++j) {
      const int col = col0 + j * 16 + nl;
      if (EPI == EPI_VT) {
        // store V transposed per head: Vt[b][h][d][t] (bf16), t contiguous in r
        const int b  = rowb / TSEQ;
        const int t0 = (rowb % TSEQ) + half * 8;
        const int h  = col / HDX;
        const int d  = col % HDX;
        v8bf pk;
#pragma unroll
        for (int r = 0; r < 8; ++r) pk[r] = (__bf16)acc[i][j][r];
        *(v8bf*)(outB + ((size_t)(b * NHEAD + h) * HDX + d) * TSEQ + t0) = pk;
      } else {
#pragma unroll
        for (int r = 0; r < 8; ++r) {
          const int row = rowb + r + half * 8;
          const size_t o = (size_t)row * N + col;
          float v = acc[i][j][r];
          if (EPI == EPI_BF16) {
            outB[o] = (__bf16)v;
          } else if (EPI == EPI_RESF32) {
            outF[o] = v + res[o];
          } else if (EPI == EPI_GELU_BF16) {
            v += bias[col];
            outB[o] = (__bf16)gelu_f(v);
          } else {  // EPI_BIAS_RES_F32
            outF[o] = v + bias[col] + res[o];
          }
        }
      }
    }
  }
}

// ---------------------------------------------------------------------------
// Causal flash attention. One wave per 16-query tile, 4 waves/block.
// Q,K row-major bf16 [N,DIMX]; V pre-transposed bf16 [B*NH][HDX][TSEQ].
// Wave-uniform fast path skips causal masking for fully-unmasked kv steps.
// ---------------------------------------------------------------------------
__global__ __launch_bounds__(128) void attn_k(const __bf16* __restrict__ Q,
                                              const __bf16* __restrict__ Km,
                                              const __bf16* __restrict__ Vt,
                                              __bf16* __restrict__ O) {
  __shared__ __bf16 lds[4][16 * 32];  // per-wave P tile: [16 q][32 kv]
  const int wave = threadIdx.x >> 5;
  const int lane = threadIdx.x & 31;
  const int half = lane >> 4;
  const int nl   = lane & 15;
  const int bh   = blockIdx.x;
  const int b    = bh / NHEAD;
  const int h    = bh - b * NHEAD;
  const int q0   = blockIdx.y * 64 + wave * 16;

  union AF { v16bf v; v8bf p[2]; };
  AF qa[2];
  const size_t qrow = (size_t)(b * TSEQ + q0 + nl) * DIMX + h * HDX;
#pragma unroll
  for (int ks = 0; ks < 2; ++ks) {
    qa[ks].p[0] = *(const v8bf*)(Q + qrow + ks * 32 + half * 16);
    qa[ks].p[1] = *(const v8bf*)(Q + qrow + ks * 32 + half * 16 + 8);
  }
  const v8f zero = {};
  v8f oc[4] = {zero, zero, zero, zero};
  float mrow[8], lrow[8];
#pragma unroll
  for (int r = 0; r < 8; ++r) { mrow[r] = -1e30f; lrow[r] = 0.f; }

  const int kvEnd = q0 + 16;  // causal: keys <= query
  const size_t kbase = (size_t)(b * TSEQ) * DIMX + h * HDX;
  __bf16* pl = &lds[wave][0];

  for (int kb = 0; kb < kvEnd; kb += 32) {
    const bool have2 = (kb + 16 < kvEnd);  // upper half-tile has any valid kv
    // --- scores S = Q K^T ---
    v8f sc[2];
    sc[0] = zero;
    sc[1] = zero;
    {
      const size_t krow = kbase + (size_t)(kb + nl) * DIMX + half * 16;
#pragma unroll
      for (int ks = 0; ks < 2; ++ks) {
        v16bf kf = *(const v16bf*)(Km + krow + ks * 32);
        sc[0] = wmma_bf16(qa[ks].v, kf, sc[0]);
      }
    }
    if (have2) {
      const size_t krow = kbase + (size_t)(kb + 16 + nl) * DIMX + half * 16;
#pragma unroll
      for (int ks = 0; ks < 2; ++ks) {
        v16bf kf = *(const v16bf*)(Km + krow + ks * 32);
        sc[1] = wmma_bf16(qa[ks].v, kf, sc[1]);
      }
    }
    // --- scale, causal mask (skipped on fully-unmasked steps), row max ---
    float s[2][8], rm[8];
    if (kb + 31 <= q0) {  // wave-uniform: whole 32-kv step below diagonal
#pragma unroll
      for (int r = 0; r < 8; ++r) {
        s[0][r] = sc[0][r] * 0.125f;
        s[1][r] = sc[1][r] * 0.125f;
        float m2 = fmaxf(s[0][r], s[1][r]);
#pragma unroll
        for (int o = 1; o < 16; o <<= 1) m2 = fmaxf(m2, __shfl_xor(m2, o, 32));
        rm[r] = m2;
      }
    } else {
#pragma unroll
      for (int r = 0; r < 8; ++r) {
        const int qi = q0 + r + half * 8;
#pragma unroll
        for (int jj = 0; jj < 2; ++jj) {
          const int kv = kb + jj * 16 + nl;
          const float v = sc[jj][r] * 0.125f;  // 1/sqrt(64)
          s[jj][r] = (kv <= qi) ? v : -1e30f;
        }
        float m2 = fmaxf(s[0][r], s[1][r]);
#pragma unroll
        for (int o = 1; o < 16; o <<= 1) m2 = fmaxf(m2, __shfl_xor(m2, o, 32));
        rm[r] = m2;
      }
    }
    // --- online softmax update; stage P (bf16) into LDS in [q][kv] order ---
#pragma unroll
    for (int r = 0; r < 8; ++r) {
      const float mn = fmaxf(mrow[r], rm[r]);
      const float al = __expf(mrow[r] - mn);
      mrow[r] = mn;
      const float p0 = __expf(s[0][r] - mn);
      const float p1 = __expf(s[1][r] - mn);
      pl[(r + half * 8) * 32 + nl]      = (__bf16)p0;
      pl[(r + half * 8) * 32 + 16 + nl] = (__bf16)p1;
      float rs = p0 + p1;
#pragma unroll
      for (int o = 1; o < 16; o <<= 1) rs += __shfl_xor(rs, o, 32);
      lrow[r] = lrow[r] * al + rs;
#pragma unroll
      for (int d = 0; d < 4; ++d) oc[d][r] *= al;
    }
    // --- reload P as A-fragment (16x32), O += P @ V ---
    AF pa;
    pa.p[0] = *(const v8bf*)(pl + nl * 32 + half * 16);
    pa.p[1] = *(const v8bf*)(pl + nl * 32 + 8 + half * 16);
    const size_t vbase = (size_t)bh * HDX * TSEQ + kb + half * 16;
#pragma unroll
    for (int d = 0; d < 4; ++d) {
      v16bf vf = *(const v16bf*)(Vt + vbase + (size_t)(d * 16 + nl) * TSEQ);
      oc[d] = wmma_bf16(pa.v, vf, oc[d]);
    }
  }
  // --- normalize and store O (bf16, row-major [N,DIMX]) ---
#pragma unroll
  for (int d = 0; d < 4; ++d) {
#pragma unroll
    for (int r = 0; r < 8; ++r) {
      const float ov = oc[d][r] / lrow[r];
      O[(size_t)(b * TSEQ + q0 + r + half * 8) * DIMX + h * HDX + d * 16 + nl] =
          (__bf16)ov;
    }
  }
}

// ---------------------------------------------------------------------------
extern "C" void kernel_launch(void* const* d_in, const int* in_sizes, int n_in,
                              void* d_out, int out_size, void* d_ws, size_t ws_size,
                              hipStream_t stream) {
  const float* x   = (const float*)d_in[0];
  const float* ln1 = (const float*)d_in[1];
  const float* ln2 = (const float*)d_in[2];
  const float* wq  = (const float*)d_in[3];
  const float* wk  = (const float*)d_in[4];
  const float* wv  = (const float*)d_in[5];
  const float* wo  = (const float*)d_in[6];
  const float* fw1 = (const float*)d_in[7];
  const float* fb1 = (const float*)d_in[8];
  const float* fw2 = (const float*)d_in[9];
  const float* fb2 = (const float*)d_in[10];

  constexpr size_t SZ_W384 = (size_t)DIMX * DIMX * 2;   // 294912
  constexpr size_t SZ_W768 = (size_t)DIMX * HDIMX * 2;  // 589824
  constexpr size_t SZ_ACT  = (size_t)NTOK * DIMX * 2;   // 12.58 MB bf16

  char* p = (char*)d_ws;
  __bf16* wqp = (__bf16*)(p + 0 * SZ_W384);
  __bf16* wkp = (__bf16*)(p + 1 * SZ_W384);
  __bf16* wvp = (__bf16*)(p + 2 * SZ_W384);
  __bf16* wop = (__bf16*)(p + 3 * SZ_W384);
  __bf16* w1p = (__bf16*)(p + 4 * SZ_W384);
  __bf16* w2p = (__bf16*)(p + 4 * SZ_W384 + SZ_W768);
  char* act = p + 4 * SZ_W384 + 2 * SZ_W768;
  __bf16* hbf = (__bf16*)(act + 0 * SZ_ACT);  // h, later reused as attention O
  __bf16* qbf = (__bf16*)(act + 1 * SZ_ACT);  // q, later reused as h2
  __bf16* kbf = (__bf16*)(act + 2 * SZ_ACT);  // k; [kbf..vt) later reused as ffn1
  __bf16* vt  = (__bf16*)(act + 3 * SZ_ACT);  // V transposed [B*NH][HD][T]
  float*  x1  = (float*)(act + 4 * SZ_ACT);   // residual stream after attention

  // pack weights to WMMA B-fragment layout (bf16)
  pack_w_k<<<(DIMX * DIMX + 255) / 256, 256, 0, stream>>>(wq, wqp, DIMX, DIMX);
  pack_w_k<<<(DIMX * DIMX + 255) / 256, 256, 0, stream>>>(wk, wkp, DIMX, DIMX);
  pack_w_k<<<(DIMX * DIMX + 255) / 256, 256, 0, stream>>>(wv, wvp, DIMX, DIMX);
  pack_w_k<<<(DIMX * DIMX + 255) / 256, 256, 0, stream>>>(wo, wop, DIMX, DIMX);
  pack_w_k<<<(DIMX * HDIMX + 255) / 256, 256, 0, stream>>>(fw1, w1p, DIMX, HDIMX);
  pack_w_k<<<(HDIMX * DIMX + 255) / 256, 256, 0, stream>>>(fw2, w2p, HDIMX, DIMX);

  const dim3 g384(NTOK / 256, DIMX / 64);   // (64, 6)
  const dim3 g768(NTOK / 256, HDIMX / 64);  // (64, 12)

  // h = rmsnorm(x, ln1)
  rmsnorm_k<<<NTOK / 8, 256, 0, stream>>>(x, ln1, hbf);
  // q, k, v projections (v stored transposed per head)
  gemm_wmma<EPI_BF16><<<g384, 256, 0, stream>>>(hbf, wqp, nullptr, nullptr, qbf,
                                                nullptr, NTOK, DIMX, DIMX);
  gemm_wmma<EPI_BF16><<<g384, 256, 0, stream>>>(hbf, wkp, nullptr, nullptr, kbf,
                                                nullptr, NTOK, DIMX, DIMX);
  gemm_wmma<EPI_VT><<<g384, 256, 0, stream>>>(hbf, wvp, nullptr, nullptr, vt,
                                              nullptr, NTOK, DIMX, DIMX);
  // causal attention -> O (reuse hbf)
  attn_k<<<dim3(BATCH * NHEAD, TSEQ / 64), 128, 0, stream>>>(qbf, kbf, vt, hbf);
  // x1 = x + O @ wo
  gemm_wmma<EPI_RESF32><<<g384, 256, 0, stream>>>(hbf, wop, nullptr, x, nullptr,
                                                  x1, NTOK, DIMX, DIMX);
  // h2 = rmsnorm(x1, ln2) (reuse qbf)
  rmsnorm_k<<<NTOK / 8, 256, 0, stream>>>(x1, ln2, qbf);
  // ffn1 = gelu(h2 @ w1 + b1) (reuse kbf..vt as 25 MB bf16 buffer)
  gemm_wmma<EPI_GELU_BF16><<<g768, 256, 0, stream>>>(qbf, w1p, fb1, nullptr, kbf,
                                                     nullptr, NTOK, DIMX, HDIMX);
  // out = x1 + ffn1 @ w2 + b2
  gemm_wmma<EPI_BIAS_RES_F32><<<g384, 256, 0, stream>>>(
      kbf, w2p, fb2, x1, nullptr, (float*)d_out, NTOK, HDIMX, DIMX);
}
